// mix_model_53721450938660
// MI455X (gfx1250) — matrix operand order repaired and checked by reading, not compiled, not run on hardware
//
#include <hip/hip_runtime.h>
#include <hip/hip_bf16.h>

// ---------------------------------------------------------------------------
// MI455X (gfx1250) fused two-tower dot-product model:
//   uf = user @ Wu + bu ; ef = event @ We + be ; out = sigmoid(sum(uf*ef, -1))
//
// Roofline: 34.6 MB of mandatory traffic -> ~1.5us @ 23.3 TB/s. We keep the
// matrix pipe under that floor by using split-bf16 (hi+lo) WMMA: 3x
// v_wmma_f32_16x16x32_bf16 per K=32 step gives ~fp32 accuracy at 8/3 the
// K-throughput of the fp32 16x16x4 WMMA.
// ---------------------------------------------------------------------------

typedef __attribute__((ext_vector_type(16))) __bf16 v16bf;
typedef __attribute__((ext_vector_type(8)))  float  v8f;
typedef __attribute__((ext_vector_type(4)))  float  v4f;

constexpr int N_ROWS = 8192;
constexpr int D_IN   = 512;
constexpr int D_EMB  = 256;
constexpr int KT     = D_IN  / 32;   // 16 K-tiles of depth 32
constexpr int CT     = D_EMB / 16;   // 16 column tiles of width 16
constexpr int FRAG   = 16 * 32;      // 512 bf16 elems per 16x32 fragment
constexpr int WMAT   = CT * KT * FRAG;           // elems per (matrix,precision) plane
constexpr int WAVES_PER_BLOCK = 2;

// ---------------------------------------------------------------------------
// Prep: split Wu/We (fp32) into bf16 hi/lo planes, pre-swizzled into the CDNA5
// WMMA B-fragment layout so the GEMM loop reads 32 contiguous bytes per lane.
// B (32x16, 16-bit) k order chosen here (not AMD's table order): lane l, element j -> K = 16*(l>>4) + j, N = l&15.
// ws plane order: [Wu_hi][Wu_lo][We_hi][We_lo], each WMAT bf16 elems.
// ---------------------------------------------------------------------------
__global__ void prep_w_kernel(const float* __restrict__ Wu,
                              const float* __restrict__ We,
                              __bf16* __restrict__ ws) {
  const int wid  = (blockIdx.x * blockDim.x + threadIdx.x) >> 5;  // 0..511
  const int lane = threadIdx.x & 31;
  const int mat  = wid >> 8;           // 0 = Wu, 1 = We
  const int ct   = (wid >> 4) & 15;
  const int kt   = wid & 15;

  const float* W = mat ? We : Wu;
  const int n   = ct * 16 + (lane & 15);
  const int kbb = (lane >> 4) * 16;

  v16bf hi, lo;
#pragma unroll
  for (int j = 0; j < 16; ++j) {
    const int k = kt * 32 + kbb + j;
    const float x = W[k * D_EMB + n];
    const __bf16 h = (__bf16)x;
    hi[j] = h;
    lo[j] = (__bf16)(x - (float)h);
  }
  __bf16* hbase = ws + mat * (2 * WMAT);
  const int off = (ct * KT + kt) * FRAG + lane * 16;
  *(v16bf*)(hbase + off)        = hi;
  *(v16bf*)(hbase + WMAT + off) = lo;
}

// ---------------------------------------------------------------------------
// Main: one wave32 per 16-row tile.
// Phase 1: stage split-bf16 A fragments (user & event rows) in LDS using the
//   same contiguous k order as B (not AMD's table order): lane l, elem j -> K = 16*(l>>4) + j,
//   M = l&15. 64 KB LDS per wave (320 KB/WGP makes this cheap).
// Phase 2: per column tile, accumulate uf/ef 16x16 tiles over K=512 with
//   3 bf16 WMMAs each (hi*hi + hi*lo + lo*hi), fuse bias + elementwise
//   product into 8 per-lane partials, butterfly-reduce over the N lanes.
// ---------------------------------------------------------------------------
__global__ void fused_two_tower_kernel(const float* __restrict__ user,
                                       const float* __restrict__ event,
                                       const float* __restrict__ bu,
                                       const float* __restrict__ be,
                                       const __bf16* __restrict__ wfrag,
                                       float* __restrict__ out) {
  extern __shared__ __bf16 sm[];

  const int lane = threadIdx.x & 31;
  const int wv   = threadIdx.x >> 5;
  const int tile = blockIdx.x * WAVES_PER_BLOCK + wv;   // 0..511
  const int m0   = tile * 16;
  const int m    = lane & 15;
  const int kb   = (lane >> 4) * 16;

  __bf16* su_hi = sm + wv * (4 * KT * FRAG);
  __bf16* su_lo = su_hi + KT * FRAG;
  __bf16* se_hi = su_lo + KT * FRAG;
  __bf16* se_lo = se_hi + KT * FRAG;

  const float* urow = user  + (size_t)(m0 + m) * D_IN;
  const float* erow = event + (size_t)(m0 + m) * D_IN;

  // ---- Phase 1: stage split-bf16 A fragments in LDS ----
#pragma unroll 1
  for (int kt = 0; kt < KT; ++kt) {
    const int k0 = kt * 32 + kb;
    float ub[16], eb[16];
    *(v4f*)(ub + 0)  = *(const v4f*)(urow + k0);
    *(v4f*)(ub + 4)  = *(const v4f*)(urow + k0 + 4);
    *(v4f*)(ub + 8)  = *(const v4f*)(urow + k0 + 8);
    *(v4f*)(ub + 12) = *(const v4f*)(urow + k0 + 12);
    *(v4f*)(eb + 0)  = *(const v4f*)(erow + k0);
    *(v4f*)(eb + 4)  = *(const v4f*)(erow + k0 + 4);
    *(v4f*)(eb + 8)  = *(const v4f*)(erow + k0 + 8);
    *(v4f*)(eb + 12) = *(const v4f*)(erow + k0 + 12);

    v16bf uh, ul, eh, el;
#pragma unroll
    for (int j = 0; j < 16; ++j) {
      const float xu = ub[j];
      const __bf16 hu = (__bf16)xu;
      uh[j] = hu;
      ul[j] = (__bf16)(xu - (float)hu);
      const float xe = eb[j];
      const __bf16 he = (__bf16)xe;
      eh[j] = he;
      el[j] = (__bf16)(xe - (float)he);
    }
    const int off = kt * FRAG + lane * 16;
    *(v16bf*)(su_hi + off) = uh;
    *(v16bf*)(su_lo + off) = ul;
    *(v16bf*)(se_hi + off) = eh;
    *(v16bf*)(se_lo + off) = el;
  }
  __syncthreads();

  const __bf16* wu_hi = wfrag;
  const __bf16* wu_lo = wfrag + WMAT;
  const __bf16* we_hi = wfrag + 2 * WMAT;
  const __bf16* we_lo = wfrag + 3 * WMAT;

  float p[8];
#pragma unroll
  for (int r = 0; r < 8; ++r) p[r] = 0.0f;

  // ---- Phase 2: WMMA GEMM tiles + fused dot product ----
#pragma unroll 1
  for (int ct = 0; ct < CT; ++ct) {
    v8f au = {};
    v8f ae = {};
#pragma unroll 2
    for (int kt = 0; kt < KT; ++kt) {
      const int aoff = kt * FRAG + lane * 16;
      const v16bf uh = *(const v16bf*)(su_hi + aoff);
      const v16bf ul = *(const v16bf*)(su_lo + aoff);
      const v16bf eh = *(const v16bf*)(se_hi + aoff);
      const v16bf el = *(const v16bf*)(se_lo + aoff);

      const int boff = (ct * KT + kt) * FRAG + lane * 16;
      const v16bf buh = *(const v16bf*)(wu_hi + boff);
      const v16bf bul = *(const v16bf*)(wu_lo + boff);
      const v16bf beh = *(const v16bf*)(we_hi + boff);
      const v16bf bel = *(const v16bf*)(we_lo + boff);

      // user tower: hi*hi + hi*lo + lo*hi
      au = __builtin_amdgcn_wmma_f32_16x16x32_bf16(false, uh, false, buh,
                                                   (short)0, au, false, false);
      au = __builtin_amdgcn_wmma_f32_16x16x32_bf16(false, uh, false, bul,
                                                   (short)0, au, false, false);
      au = __builtin_amdgcn_wmma_f32_16x16x32_bf16(false, ul, false, buh,
                                                   (short)0, au, false, false);
      // event tower
      ae = __builtin_amdgcn_wmma_f32_16x16x32_bf16(false, eh, false, beh,
                                                   (short)0, ae, false, false);
      ae = __builtin_amdgcn_wmma_f32_16x16x32_bf16(false, eh, false, bel,
                                                   (short)0, ae, false, false);
      ae = __builtin_amdgcn_wmma_f32_16x16x32_bf16(false, el, false, beh,
                                                   (short)0, ae, false, false);
    }
    // C/D layout: lane holds column N = ct*16 + (lane&15) for rows
    // M = r + 8*(lane>>4) -> bias is a per-lane scalar across all 8 rows.
    const float bus = bu[ct * 16 + m];
    const float bes = be[ct * 16 + m];
#pragma unroll
    for (int r = 0; r < 8; ++r) p[r] += (au[r] + bus) * (ae[r] + bes);
  }

  // ---- Reduce over N (16 lanes per half-wave), sigmoid, store ----
#pragma unroll
  for (int r = 0; r < 8; ++r) {
    float v = p[r];
    v += __shfl_xor(v, 1, 32);
    v += __shfl_xor(v, 2, 32);
    v += __shfl_xor(v, 4, 32);
    v += __shfl_xor(v, 8, 32);
    p[r] = v;
  }
  if ((lane & 15) == 0) {
    const int rbase = m0 + (lane >> 4) * 8;  // lane0 -> rows 0..7, lane16 -> 8..15
    v4f o0, o1;
#pragma unroll
    for (int r = 0; r < 4; ++r) {
      o0[r] = 1.0f / (1.0f + __expf(-p[r]));
      o1[r] = 1.0f / (1.0f + __expf(-p[r + 4]));
    }
    *(v4f*)(out + rbase)     = o0;
    *(v4f*)(out + rbase + 4) = o1;
  }
}

// ---------------------------------------------------------------------------
extern "C" void kernel_launch(void* const* d_in, const int* in_sizes, int n_in,
                              void* d_out, int out_size, void* d_ws, size_t ws_size,
                              hipStream_t stream) {
  const float* user  = (const float*)d_in[0];
  const float* event = (const float*)d_in[1];
  const float* Wu    = (const float*)d_in[2];
  const float* bu    = (const float*)d_in[3];
  const float* We    = (const float*)d_in[4];
  const float* be    = (const float*)d_in[5];
  float*   out = (float*)d_out;
  __bf16*  ws  = (__bf16*)d_ws;   // needs 4*WMAT*2 bytes = 1 MB

  // 512 weight fragments (2 matrices x 16 ct x 16 kt), one wave each.
  prep_w_kernel<<<128, 128, 0, stream>>>(Wu, We, ws);

  const int blocks = (N_ROWS / 16) / WAVES_PER_BLOCK;                 // 256
  const size_t shmem = (size_t)WAVES_PER_BLOCK * 4 * KT * FRAG * 2;   // 128 KB
  fused_two_tower_kernel<<<blocks, WAVES_PER_BLOCK * 32, shmem, stream>>>(
      user, event, bu, be, ws, out);
}
